// GraphSAGE_9466107921073
// MI455X (gfx1250) — matrix operand order repaired and checked by reading, not compiled, not run on hardware
//
#include <hip/hip_runtime.h>
#include <hip/hip_bf16.h>
#include <math.h>

typedef __attribute__((ext_vector_type(16))) _Float16 v16h;
typedef __attribute__((ext_vector_type(8)))  float    v8f;

// ---------------------------------------------------------------------------
// Degree count: one thread per edge, atomic f32 add into cnt[dst].
// ---------------------------------------------------------------------------
__global__ void degree_kernel(const int* __restrict__ dst, float* __restrict__ cnt, int E) {
    int e = blockIdx.x * blockDim.x + threadIdx.x;
    if (e < E) atomicAdd(&cnt[dst[e]], 1.0f);
}

__global__ void inv_kernel(float* __restrict__ cnt, int N) {
    int i = blockIdx.x * blockDim.x + threadIdx.x;
    if (i < N) cnt[i] = 1.0f / fmaxf(cnt[i], 1.0f);
}

// ---------------------------------------------------------------------------
// Edge aggregation: 32 lanes of a wave cover one edge's 128-float feature row
// (float4 per lane) -> coalesced 512B gather from h[src], 4 global f32
// atomics per lane into agg[dst]. agg must be pre-zeroed.
// ---------------------------------------------------------------------------
__global__ void agg_kernel(const float* __restrict__ h, const int* __restrict__ src,
                           const int* __restrict__ dst, float* __restrict__ agg, int E) {
    int idx = blockIdx.x * blockDim.x + threadIdx.x;
    int e = idx >> 5;                    // 32 chunks x 4 floats = 128 features
    if (e >= E) return;
    int c = (idx & 31) << 2;
    int s = src[e], d = dst[e];
    const float4 v = *(const float4*)(h + (size_t)s * 128 + c);
    float* o = agg + (size_t)d * 128 + c;
    atomicAdd(o + 0, v.x);
    atomicAdd(o + 1, v.y);
    atomicAdd(o + 2, v.z);
    atomicAdd(o + 3, v.w);
}

// ---------------------------------------------------------------------------
// Fused SAGE layer: HOUT = sigmoid( (AGG*inv_cnt) @ Wl^T + b + HIN @ Wr^T )
// using v_wmma_f32_16x16x32_f16. One wave = one 16-row output tile, looping
// over 4 column tiles (64 cols per block) so A fragments are reused 4x.
// Weights converted to f16 in LDS once per block. If REDUCE, the layer output
// is column-summed (LDS ds_add_f32 then one global atomic per column/block)
// instead of stored.
//
// Register layouts per CDNA5 ISA (hs = lane>>4, ln = lane&15):
//   A[i]  : row = ln,            K = (i&7) + ((i>>3)<<4) + (hs<<3)
//   B[i]  : col = ln,            K = (i&7) + ((i>>3)<<4) + (hs<<3)
//   D[r]  : row = r + (hs<<3),   col = ln
// ---------------------------------------------------------------------------
template <int DIN, int DOUT, bool REDUCE>
__global__ void __launch_bounds__(256)
sage_layer(const float* __restrict__ AGG, const float* __restrict__ HIN,
           const float* __restrict__ invc, const float* __restrict__ Wl,
           const float* __restrict__ bias, const float* __restrict__ Wr,
           float* __restrict__ HOUT, float* __restrict__ OUTSUM, int Nnodes) {
    constexpr int NTB  = 4;              // 16x16 column tiles per block
    constexpr int COLS = NTB * 16;       // 64 output columns per block
    __shared__ _Float16 sWl[COLS * DIN]; // 16 KB
    __shared__ _Float16 sWr[COLS * DIN]; // 16 KB
    __shared__ float    red[REDUCE ? COLS : 1];

    const int colBase = blockIdx.y * COLS;
    // stage weights (row-major [DOUT][DIN]) into LDS as f16
    for (int i = threadIdx.x; i < COLS * DIN; i += 256) {
        sWl[i] = (_Float16)Wl[(size_t)colBase * DIN + i];
        sWr[i] = (_Float16)Wr[(size_t)colBase * DIN + i];
    }
    if (REDUCE)
        for (int i = threadIdx.x; i < COLS; i += 256) red[i] = 0.0f;
    __syncthreads();

    const int wave = threadIdx.x >> 5;   // wave32
    const int lane = threadIdx.x & 31;
    const int hs   = lane >> 4;
    const int ln   = lane & 15;
    const int rowTile = blockIdx.x * 8 + wave;
    const bool active = (rowTile * 16) < Nnodes;   // wave-uniform: EXEC stays full

    if (active) {
        const int   row0 = rowTile * 16;
        const int   m    = row0 + ln;              // A-fragment row for this lane
        const float ic   = invc[m];

        v8f acc[NTB] = {};
        for (int kb = 0; kb < DIN; kb += 32) {
            v16h a1, a2;
#pragma unroll
            for (int i = 0; i < 16; ++i) {
                const int kl = (i & 7) + ((i >> 3) << 4) + (hs << 3);
                const int k  = kb + kl;
                a1[i] = (_Float16)(AGG[(size_t)m * DIN + k] * ic);  // mean aggregate
                a2[i] = (_Float16)(HIN[(size_t)m * DIN + k]);       // self features
            }
#pragma unroll
            for (int t = 0; t < NTB; ++t) {
                v16h b1, b2;
#pragma unroll
                for (int i = 0; i < 16; ++i) {
                    const int k = kb + (i & 7) + ((i >> 3) << 4) + (hs << 3);
                    b1[i] = sWl[(t * 16 + ln) * DIN + k];   // B[k][n] = W[n][k]
                    b2[i] = sWr[(t * 16 + ln) * DIN + k];
                }
                acc[t] = __builtin_amdgcn_wmma_f32_16x16x32_f16(
                    false, a1, false, b1, (short)0, acc[t], false, false);
                acc[t] = __builtin_amdgcn_wmma_f32_16x16x32_f16(
                    false, a2, false, b2, (short)0, acc[t], false, false);
            }
        }

#pragma unroll
        for (int t = 0; t < NTB; ++t) {
            const int col = colBase + t * 16 + ln;
            const float bv = bias[col];
            if (!REDUCE) {
#pragma unroll
                for (int r = 0; r < 8; ++r) {
                    const int row = row0 + r + (hs << 3);
                    float v = acc[t][r] + bv;
                    v = 1.0f / (1.0f + __expf(-v));
                    HOUT[(size_t)row * DOUT + col] = v;
                }
            } else {
                float s = 0.0f;
#pragma unroll
                for (int r = 0; r < 8; ++r) {
                    float v = acc[t][r] + bv;
                    s += 1.0f / (1.0f + __expf(-v));
                }
                atomicAdd(&red[t * 16 + ln], s);   // ds_add_f32, block-local
            }
        }
    }

    if (REDUCE) {
        __syncthreads();
        for (int i = threadIdx.x; i < COLS; i += 256)
            atomicAdd(&OUTSUM[colBase + i], red[i]);   // 1 global atomic / col / block
    }
}

// ---------------------------------------------------------------------------
// Host-side orchestration (graph-capture safe: only memsets + launches).
// ---------------------------------------------------------------------------
extern "C" void kernel_launch(void* const* d_in, const int* in_sizes, int n_in,
                              void* d_out, int out_size, void* d_ws, size_t ws_size,
                              hipStream_t stream) {
    const float* x   = (const float*)d_in[0];
    const int*   ei  = (const int*)d_in[1];    // harness passes integers as int32
    const float* W1l = (const float*)d_in[2];
    const float* b1  = (const float*)d_in[3];
    const float* W1r = (const float*)d_in[4];
    const float* W2l = (const float*)d_in[5];
    const float* b2  = (const float*)d_in[6];
    const float* W2r = (const float*)d_in[7];
    const float* W3l = (const float*)d_in[8];
    const float* b3  = (const float*)d_in[9];
    const float* W3r = (const float*)d_in[10];

    const int N = in_sizes[0] / 128;
    const int E = in_sizes[1] / 2;
    const int* src = ei;
    const int* dst = ei + E;

    // workspace: cnt/inv [N] | AGG [N*128] | H1 [N*128] | H2 [N*128]  (~74 MB)
    float* ws  = (float*)d_ws;
    float* cnt = ws;
    float* agg = cnt + N;
    float* h1  = agg + (size_t)N * 128;
    float* h2  = h1  + (size_t)N * 128;
    float* out = (float*)d_out;

    // degrees (reused by all layers)
    hipMemsetAsync(cnt, 0, (size_t)N * sizeof(float), stream);
    degree_kernel<<<(E + 255) / 256, 256, 0, stream>>>(dst, cnt, E);
    inv_kernel<<<(N + 255) / 256, 256, 0, stream>>>(cnt, N);

    const int rowTiles = (N + 15) / 16;
    const int gx = (rowTiles + 7) / 8;                       // 8 waves/block, 128 rows/block
    const unsigned aggBlocks = ((unsigned)E * 32u + 255u) / 256u;

    // layer 1: x -> h1
    hipMemsetAsync(agg, 0, (size_t)N * 128 * sizeof(float), stream);
    agg_kernel<<<aggBlocks, 256, 0, stream>>>(x, src, dst, agg, E);
    sage_layer<128, 128, false><<<dim3(gx, 2), 256, 0, stream>>>(
        agg, x, cnt, W1l, b1, W1r, h1, nullptr, N);

    // layer 2: h1 -> h2
    hipMemsetAsync(agg, 0, (size_t)N * 128 * sizeof(float), stream);
    agg_kernel<<<aggBlocks, 256, 0, stream>>>(h1, src, dst, agg, E);
    sage_layer<128, 128, false><<<dim3(gx, 2), 256, 0, stream>>>(
        agg, h1, cnt, W2l, b2, W2r, h2, nullptr, N);

    // layer 3 fused with final column-sum: h2 -> d_out[64]
    hipMemsetAsync(agg, 0, (size_t)N * 128 * sizeof(float), stream);
    agg_kernel<<<aggBlocks, 256, 0, stream>>>(h2, src, dst, agg, E);
    hipMemsetAsync(out, 0, (size_t)out_size * sizeof(float), stream);
    sage_layer<128, 64, true><<<dim3(gx, 1), 256, 0, stream>>>(
        agg, h2, cnt, W3l, b3, W3r, nullptr, out, N);
}
